// TransformerDistancePredictor_39462159516219
// MI455X (gfx1250) — matrix operand order repaired and checked by reading, not compiled, not run on hardware
//
#include <hip/hip_runtime.h>
#include <hip/hip_bf16.h>

typedef __bf16 bh16;
typedef __attribute__((ext_vector_type(8)))  float v8f;
typedef __attribute__((ext_vector_type(8)))  bh16  v8bf;
typedef __attribute__((ext_vector_type(16))) bh16  v16bf;

#define NLAY 4
#define HD   256
#define PDIM 64
#define NHEAD 4
#define NBINS 65
#define BB   2
#define LSEQ 384
#define PROWS (BB*LSEQ*LSEQ)   // 294912

__device__ __forceinline__ float geluf(float x){
  return 0.5f*x*(1.0f+erff(x*0.70710678118654752f));
}

// index into WMMA B-operand packed layout (K-major 32-blocks, N 16-blocks)
__device__ __forceinline__ int bidx(int k,int n,int NTtot){
  int kt=k>>5, e=(k&7)|(((k>>4)&1)<<3), lh=(k>>3)&1, nt=n>>4, ll=n&15;
  return ((kt*NTtot+nt)*32 + lh*16+ll)*16 + e;
}

__device__ __forceinline__ v16bf mk_a(v8bf lo, v8bf hi){
  v16bf a;
#pragma unroll
  for(int e=0;e<8;e++){ a[e]=lo[e]; a[8+e]=hi[e]; }
  return a;
}

// ---------------------------------------------------------------- weight pack
__global__ void pack_w(const float* __restrict__ W, bh16* __restrict__ P,
                       int K, int N, int NTtot){
  int idx = blockIdx.x*256+threadIdx.x;
  int total = (K>>5)*NTtot*512;
  if(idx>=total) return;
  int e=idx&15, lane=(idx>>4)&31, nt=(idx>>9)%NTtot, kt=idx/(512*NTtot);
  int k = kt*32 + ((lane>>4)<<3) + (e&7) + ((e>>3)<<4);
  int n = nt*16 + (lane&15);
  P[idx] = (bh16)((n<N)? W[(long)k*N+n] : 0.0f);
}

// wsum[i][c] = sum_h pb_w[i,c,h] / (B*NH);  pad dh_b2 to 80
__global__ void prep_small(const float* __restrict__ pb_w, const float* __restrict__ dh_b2,
                           float* __restrict__ wsum, float* __restrict__ dh_b2p){
  int t=threadIdx.x;
  if(t<NLAY*PDIM){
    const float* p = pb_w + (long)t*NHEAD;
    wsum[t]=(p[0]+p[1]+p[2]+p[3])*(1.0f/(BB*NHEAD));
  }
  if(t<80) dh_b2p[t] = (t<NBINS)? dh_b2[t] : 0.0f;
}

// ---------------------------------------------------------------- seq init
__global__ void seq_init(const float* __restrict__ x, const float* __restrict__ rp_w,
                         const float* __restrict__ rp_b, const float* __restrict__ pos,
                         float* __restrict__ h, bh16* __restrict__ hbf){
  int idx = blockIdx.x*256+threadIdx.x; if(idx>=BB*LSEQ*HD) return;
  int row = idx>>8, c = idx&255, l = row%LSEQ;
  float s = rp_b[c] + pos[l*HD+c];
  const float* xr = x + (long)row*48;
#pragma unroll 8
  for(int k=0;k<48;k++) s += xr[k]*rp_w[k*HD+c];
  h[idx]=s; hbf[idx]=(bh16)s;
}

__global__ void pair_init(const float* __restrict__ pi, const float* __restrict__ pj,
                          const float* __restrict__ rel_emb, float* __restrict__ pair){
  long idx = (long)blockIdx.x*256+threadIdx.x;
  if(idx >= (long)PROWS*16) return;
  long p = idx>>4; int c = (int)(idx&15)*4;
  int bb = (int)(p/(LSEQ*LSEQ)); int rem=(int)(p%(LSEQ*LSEQ));
  int ii = rem/LSEQ, jj = rem%LSEQ;
  int rel = jj-ii; rel = rel<-32?-32:(rel>32?32:rel); rel += 32;
  float4 a=*(const float4*)(pi+((long)(bb*LSEQ+ii))*PDIM+c);
  float4 b=*(const float4*)(pj+((long)(bb*LSEQ+jj))*PDIM+c);
  float4 r=*(const float4*)(rel_emb+(long)rel*PDIM+c);
  float4 o; o.x=a.x+b.x+r.x; o.y=a.y+b.y+r.y; o.z=a.z+b.z+r.z; o.w=a.w+b.w+r.w;
  *(float4*)(pair+p*PDIM+c)=o;
}

// ---------------------------------------------------------------- LN (wave/row)
__global__ __launch_bounds__(128)
void ln_rows(const float* __restrict__ X, const float* __restrict__ g,
             const float* __restrict__ bta, bh16* __restrict__ Y, int C){
  int row = blockIdx.x*4+threadIdx.y, lane = threadIdx.x;
  const float* x = X + (long)row*C;
  float s=0.f, s2=0.f;
  for(int c=lane;c<C;c+=32){ float v=x[c]; s+=v; s2+=v*v; }
#pragma unroll
  for(int o=16;o>0;o>>=1){ s+=__shfl_xor(s,o,32); s2+=__shfl_xor(s2,o,32); }
  float mean=s/C, var=s2/C-mean*mean, rstd=rsqrtf(var+1e-5f);
  bh16* y = Y+(long)row*C;
  for(int c=lane;c<C;c+=32) y[c]=(bh16)((x[c]-mean)*rstd*g[c]+bta[c]);
}

// ---------------------------------------------------------------- pair bias mask
__global__ void mask_kernel(const float* __restrict__ pair, const float* __restrict__ wsum,
                            const float* __restrict__ g, const float* __restrict__ bb,
                            float* __restrict__ mask){
  int idx = blockIdx.x*256+threadIdx.x; if(idx>=LSEQ*LSEQ) return;
  float acc=0.f;
  for(int b=0;b<BB;b++){
    const float* row = pair + ((long)b*LSEQ*LSEQ + idx)*PDIM;
    float s=0.f,s2=0.f;
#pragma unroll 8
    for(int c=0;c<PDIM;c++){ float v=row[c]; s+=v; s2+=v*v; }
    float mean=s*(1.f/PDIM), var=s2*(1.f/PDIM)-mean*mean, rstd=rsqrtf(var+1e-5f);
    float d=0.f;
#pragma unroll 8
    for(int c=0;c<PDIM;c++) d += ((row[c]-mean)*rstd*g[c]+bb[c])*wsum[c];
    acc += d;
  }
  mask[idx]=acc;
}

// ---------------------------------------------------------------- attn pack
__global__ void attn_pack(const float* __restrict__ qkv, bh16* __restrict__ qa,
                          bh16* __restrict__ kpk, bh16* __restrict__ vpk){
  int idx = blockIdx.x*256+threadIdx.x; if(idx>=BB*NHEAD*LSEQ*64) return;
  int z = idx/(LSEQ*64); int rem = idx%(LSEQ*64);
  int m = rem/64, d = rem%64;
  int b = z>>2, hh = z&3;
  const float* base = qkv + ((long)(b*LSEQ+m))*(3*HD) + hh*64 + d;
  qa[idx] = (bh16)(base[0]*0.125f);                      // fold 1/sqrt(DH)
  kpk[(long)z*24576 + bidx(d, m, 24)] = (bh16)base[HD];
  vpk[(long)z*24576 + bidx(m, d, 4)]  = (bh16)base[2*HD];
}

__global__ __launch_bounds__(128)
void softmax_rows(const float* __restrict__ S, bh16* __restrict__ A){
  int row = blockIdx.x, tid = threadIdx.x;
  const float* s = S + (long)row*LSEQ;
  float v0=s[tid], v1=s[tid+128], v2=s[tid+256];
  __shared__ float red[128];
  red[tid]=fmaxf(v0,fmaxf(v1,v2)); __syncthreads();
  for(int o=64;o>0;o>>=1){ if(tid<o) red[tid]=fmaxf(red[tid],red[tid+o]); __syncthreads(); }
  float m=red[0]; __syncthreads();
  float e0=expf(v0-m), e1=expf(v1-m), e2=expf(v2-m);
  red[tid]=e0+e1+e2; __syncthreads();
  for(int o=64;o>0;o>>=1){ if(tid<o) red[tid]+=red[tid+o]; __syncthreads(); }
  float inv=1.0f/red[0];
  bh16* a = A + (long)row*LSEQ;
  a[tid]=(bh16)(e0*inv); a[tid+128]=(bh16)(e1*inv); a[tid+256]=(bh16)(e2*inv);
}

// ---------------------------------------------------------------- generic WMMA GEMM
// one wave -> 16 rows x 64 cols; block = (32,4) waves on 4 consecutive M tiles
enum { EPI_BIAS=0, EPI_RES=1, EPI_GELU_BF=2, EPI_MASK=3, EPI_ATTNO=4, EPI_RES_DUAL=5 };

template<int EPI>
__global__ __launch_bounds__(128) void gemm_wmma(
    const bh16* __restrict__ A, long sAz, int lda,
    const bh16* __restrict__ Bp, long sBz, int NTtot,
    const float* __restrict__ bias,
    float* __restrict__ Of, bh16* __restrict__ Ob, long sOz, int ldo,
    const float* __restrict__ extra, int ldex, int K)
{
  const int lane = threadIdx.x;
  const int mtile = blockIdx.y*4 + threadIdx.y;
  const int z = blockIdx.z;
  const int ng = blockIdx.x;
  const bh16* arow = A + (long)z*sAz + (long)(mtile*16 + (lane&15))*lda;
  const bh16* Bz = Bp + (long)z*sBz;
  const int kb0 = (lane>>4)*8;
  v8f acc[4] = {};
  const int nK = K>>5;
  for(int kt=0; kt<nK; ++kt){
    v8bf lo = *(const v8bf*)(arow + kt*32 + kb0);
    v8bf hi = *(const v8bf*)(arow + kt*32 + kb0 + 16);
    if (kt+1 < nK) __builtin_prefetch(arow + (kt+1)*32 + kb0, 0, 1);
    v16bf a = mk_a(lo,hi);
    const bh16* bb = Bz + ((long)(kt*NTtot + ng*4)*32 + lane)*16;
#pragma unroll
    for(int t=0;t<4;t++){
      v16bf b = *(const v16bf*)(bb + t*512);
      acc[t] = __builtin_amdgcn_wmma_f32_16x16x32_bf16(false, a, false, b,
                                                       (short)0, acc[t], false, false);
    }
  }
  long ooff;
  if (EPI==EPI_ATTNO) ooff = (long)(z/NHEAD)*LSEQ*ldo + (long)(z%NHEAD)*64;
  else                ooff = (long)z*sOz;
  const int mb = mtile*16 + 8*(lane>>4);
#pragma unroll
  for(int t=0;t<4;t++){
    int n = ng*64 + t*16 + (lane&15);
    float bv = (EPI==EPI_MASK || EPI==EPI_ATTNO) ? 0.0f : (bias ? bias[n] : 0.0f);
#pragma unroll
    for(int r=0;r<8;r++){
      int mrow = mb + r;
      float v = acc[t][r] + bv;
      long oi_ = ooff + (long)mrow*ldo + n;
      if      (EPI==EPI_BIAS)    Of[oi_] = v;
      else if (EPI==EPI_RES)     Of[oi_] += v;
      else if (EPI==EPI_GELU_BF) Ob[oi_] = (bh16)geluf(v);
      else if (EPI==EPI_MASK)    Of[oi_] = v + extra[(long)mrow*ldex + n];
      else if (EPI==EPI_ATTNO)   Ob[oi_] = (bh16)v;
      else if (EPI==EPI_RES_DUAL){ float r2 = Of[oi_] + v; Of[oi_]=r2; Ob[oi_]=(bh16)r2; }
    }
  }
}

// ---------------------------------------------------------------- fused pair MLP
// MODE 0: pair += MLP(LN(oi[i]+oj[j]))        (pair update, per layer)
// MODE 1: logits = MLP(LN(pair))              (distance head, N padded to 80)
template<int MODE>
__global__ __launch_bounds__(128) void pair_fused(
    float* __restrict__ pair, const float* __restrict__ oi, const float* __restrict__ oj,
    const float* __restrict__ lng, const float* __restrict__ lnb,
    const bh16* __restrict__ W1, const float* __restrict__ b1,
    const bh16* __restrict__ W2, const float* __restrict__ b2,
    float* __restrict__ logits)
{
  __shared__ bh16 lA[64][64];
  __shared__ bh16 lB[64][64];
  const long p0 = (long)blockIdx.x*64;
  const int tid = threadIdx.x;
  { // phase 1: build LN(input) tile in LDS (2 threads per row)
    int r = tid>>1, half = tid&1, c0 = half*32;
    long prow = p0 + r;
    float v[32]; float s=0.f, s2=0.f;
    if (MODE==0){
      int bb=(int)(prow/(LSEQ*LSEQ)); int rem=(int)(prow%(LSEQ*LSEQ));
      int ii=rem/LSEQ, jj=rem%LSEQ;
      const float* si = oi + ((long)(bb*LSEQ+ii))*PDIM + c0;   // outer_b folded into oi
      const float* sj = oj + ((long)(bb*LSEQ+jj))*PDIM + c0;
#pragma unroll
      for(int e=0;e<32;e++){ float x=si[e]+sj[e]; v[e]=x; s+=x; s2+=x*x; }
    } else {
      const float* sp = pair + prow*PDIM + c0;
#pragma unroll
      for(int e=0;e<32;e++){ float x=sp[e]; v[e]=x; s+=x; s2+=x*x; }
    }
    s  += __shfl_xor(s,1,32);
    s2 += __shfl_xor(s2,1,32);
    float mean=s*(1.f/64.f), var=s2*(1.f/64.f)-mean*mean, rstd=rsqrtf(var+1e-5f);
#pragma unroll
    for(int e=0;e<32;e++){ int c=c0+e; lA[r][c]=(bh16)((v[e]-mean)*rstd*lng[c]+lnb[c]); }
  }
  __syncthreads();
  const int lane = tid&31, wv = tid>>5;
  const int kb0 = (lane>>4)*8;
  const int ml  = wv*16 + (lane&15);
  const int mb  = wv*16 + 8*(lane>>4);
  { // phase 2: t = gelu(lA @ W1 + b1) -> LDS
    v8f acc[4] = {};
#pragma unroll
    for(int kt=0;kt<2;kt++){
      v8bf lo = *(const v8bf*)&lA[ml][kt*32+kb0];
      v8bf hi = *(const v8bf*)&lA[ml][kt*32+kb0+16];
      v16bf a = mk_a(lo,hi);
#pragma unroll
      for(int t=0;t<4;t++){
        v16bf b = *(const v16bf*)(W1 + ((long)((kt*4+t)*32+lane))*16);
        acc[t]=__builtin_amdgcn_wmma_f32_16x16x32_bf16(false,a,false,b,(short)0,acc[t],false,false);
      }
    }
#pragma unroll
    for(int t=0;t<4;t++){
      int n=t*16+(lane&15); float bv=b1[n];
#pragma unroll
      for(int r=0;r<8;r++) lB[mb+r][n]=(bh16)geluf(acc[t][r]+bv);
    }
  }
  __syncthreads();
  { // phase 3: out = lB @ W2 + b2 (+pair residual / logits)
    const int NT2 = MODE ? 5 : 4;
    v8f acc[5] = {};
#pragma unroll
    for(int kt=0;kt<2;kt++){
      v8bf lo = *(const v8bf*)&lB[ml][kt*32+kb0];
      v8bf hi = *(const v8bf*)&lB[ml][kt*32+kb0+16];
      v16bf a = mk_a(lo,hi);
#pragma unroll
      for(int t=0;t<5;t++){
        if (t<NT2){
          v16bf b = *(const v16bf*)(W2 + ((long)((kt*NT2+t)*32+lane))*16);
          acc[t]=__builtin_amdgcn_wmma_f32_16x16x32_bf16(false,a,false,b,(short)0,acc[t],false,false);
        }
      }
    }
#pragma unroll
    for(int t=0;t<5;t++){
      if (t<NT2){
        int n=t*16+(lane&15); float bv=b2[n];
#pragma unroll
        for(int r=0;r<8;r++){
          long row=p0+mb+r; float val=acc[t][r]+bv;
          if (MODE==0) pair[row*PDIM+n] += val;
          else         logits[row*80+n] = val;
        }
      }
    }
  }
}

// ---------------------------------------------------------------- symmetrize
__global__ void symmetrize(const float* __restrict__ lg, float* __restrict__ out){
  long idx=(long)blockIdx.x*256+threadIdx.x;
  if(idx >= (long)PROWS*NBINS) return;
  int k=(int)(idx%NBINS); long t=idx/NBINS;
  int jj=(int)(t%LSEQ); long t2=t/LSEQ; int ii=(int)(t2%LSEQ); int bb=(int)(t2/LSEQ);
  long pij=((long)(bb*LSEQ+ii))*LSEQ+jj;
  long pji=((long)(bb*LSEQ+jj))*LSEQ+ii;
  out[idx]=0.5f*(lg[pij*80+k]+lg[pji*80+k]);
}

// ================================================================ host
extern "C" void kernel_launch(void* const* d_in, const int* in_sizes, int n_in,
                              void* d_out, int out_size, void* d_ws, size_t ws_size,
                              hipStream_t stream) {
  const float* x       =(const float*)d_in[0];
  const float* rp_w    =(const float*)d_in[1];
  const float* rp_b    =(const float*)d_in[2];
  const float* pos     =(const float*)d_in[3];
  const float* pii_w   =(const float*)d_in[4];
  const float* pii_b   =(const float*)d_in[5];
  const float* pij_w   =(const float*)d_in[6];
  const float* pij_b   =(const float*)d_in[7];
  const float* rel_emb =(const float*)d_in[8];
  const float* ln_seq_g=(const float*)d_in[9];
  const float* ln_seq_b=(const float*)d_in[10];
  const float* ln_pair_g=(const float*)d_in[11];
  const float* ln_pair_b=(const float*)d_in[12];
  const float* pb_w    =(const float*)d_in[13];
  const float* in_w    =(const float*)d_in[14];
  const float* in_b    =(const float*)d_in[15];
  const float* out_w   =(const float*)d_in[16];
  const float* out_b   =(const float*)d_in[17];
  const float* ff_ln_g =(const float*)d_in[18];
  const float* ff_ln_b =(const float*)d_in[19];
  const float* ff_w1   =(const float*)d_in[20];
  const float* ff_b1   =(const float*)d_in[21];
  const float* ff_w2   =(const float*)d_in[22];
  const float* ff_b2   =(const float*)d_in[23];
  const float* pu_ln_g =(const float*)d_in[24];
  const float* pu_ln_b =(const float*)d_in[25];
  const float* pu_w1   =(const float*)d_in[26];
  const float* pu_b1   =(const float*)d_in[27];
  const float* pu_w2   =(const float*)d_in[28];
  const float* pu_b2   =(const float*)d_in[29];
  const float* outer_w =(const float*)d_in[30];
  const float* outer_b =(const float*)d_in[31];
  const float* dh_ln_g =(const float*)d_in[32];
  const float* dh_ln_b =(const float*)d_in[33];
  const float* dh_w1   =(const float*)d_in[34];
  const float* dh_b1   =(const float*)d_in[35];
  const float* dh_w2   =(const float*)d_in[36];
  const float* dh_b2   =(const float*)d_in[37];

  // -------- workspace carve-up
  size_t off = 0;
  auto carve = [&](size_t bytes)->char*{
    char* p = (char*)d_ws + off; off += (bytes+255)&~(size_t)255; return p;
  };
  float* pairb  = (float*)carve((size_t)PROWS*PDIM*4);
  float* logitb = (float*)carve((size_t)PROWS*80*4);
  float* hbuf   = (float*)carve((size_t)768*HD*4);
  bh16*  hbf    = (bh16* )carve((size_t)768*HD*2);
  bh16*  seqn   = (bh16* )carve((size_t)768*HD*2);
  bh16*  ffln   = (bh16* )carve((size_t)768*HD*2);
  bh16*  ffa    = (bh16* )carve((size_t)768*1024*2);
  float* qkvb   = (float*)carve((size_t)768*768*4);
  bh16*  qa     = (bh16* )carve((size_t)8*LSEQ*64*2);
  bh16*  kpk    = (bh16* )carve((size_t)8*24576*2);
  bh16*  vpk    = (bh16* )carve((size_t)8*24576*2);
  float* scor   = (float*)carve((size_t)8*LSEQ*LSEQ*4);
  bh16*  attn   = (bh16* )carve((size_t)8*LSEQ*LSEQ*2);
  bh16*  obf    = (bh16* )carve((size_t)768*HD*2);
  float* pib    = (float*)carve((size_t)768*PDIM*4);
  float* pjb    = (float*)carve((size_t)768*PDIM*4);
  float* oib    = (float*)carve((size_t)768*PDIM*4);
  float* ojb    = (float*)carve((size_t)768*PDIM*4);
  float* maskb  = (float*)carve((size_t)LSEQ*LSEQ*4);
  float* wsum   = (float*)carve(NLAY*PDIM*4);
  float* dh_b2p = (float*)carve(80*4);
  bh16* inw_pk  = (bh16*)carve((size_t)NLAY*196608*2);
  bh16* outw_pk = (bh16*)carve((size_t)NLAY*65536*2);
  bh16* ffw1_pk = (bh16*)carve((size_t)NLAY*262144*2);
  bh16* ffw2_pk = (bh16*)carve((size_t)NLAY*262144*2);
  bh16* outA_pk = (bh16*)carve((size_t)NLAY*16384*2);
  bh16* outB_pk = (bh16*)carve((size_t)NLAY*16384*2);
  bh16* puw1_pk = (bh16*)carve((size_t)NLAY*4096*2);
  bh16* puw2_pk = (bh16*)carve((size_t)NLAY*4096*2);
  bh16* pii_pk  = (bh16*)carve((size_t)16384*2);
  bh16* pij_pk  = (bh16*)carve((size_t)16384*2);
  bh16* dhw1_pk = (bh16*)carve((size_t)4096*2);
  bh16* dhw2_pk = (bh16*)carve((size_t)64*80*2);
  (void)ws_size; (void)n_in; (void)in_sizes; (void)out_size;

  dim3 blk(32,4);
  auto packK = [&](const float* W, bh16* P, int K, int N, int Npad){
    int NTtot = Npad/16; int total=(K/32)*NTtot*512;
    pack_w<<<(total+255)/256,256,0,stream>>>(W,P,K,N,NTtot);
  };

  // -------- one-time prep (re-run each call; cheap)
  for(int i=0;i<NLAY;i++){
    packK(in_w + (long)i*HD*768,   inw_pk  + (long)i*196608, HD, 768, 768);
    packK(out_w+ (long)i*HD*HD,    outw_pk + (long)i*65536,  HD, HD,  HD);
    packK(ff_w1+ (long)i*HD*1024,  ffw1_pk + (long)i*262144, HD, 1024,1024);
    packK(ff_w2+ (long)i*1024*HD,  ffw2_pk + (long)i*262144, 1024, HD, HD);
    packK(outer_w+(long)i*2*HD*PDIM,            outA_pk+(long)i*16384, HD, PDIM, PDIM);
    packK(outer_w+(long)i*2*HD*PDIM + HD*PDIM,  outB_pk+(long)i*16384, HD, PDIM, PDIM);
    packK(pu_w1+ (long)i*PDIM*PDIM, puw1_pk+(long)i*4096, PDIM, PDIM, PDIM);
    packK(pu_w2+ (long)i*PDIM*PDIM, puw2_pk+(long)i*4096, PDIM, PDIM, PDIM);
  }
  packK(pii_w, pii_pk, HD, PDIM, PDIM);
  packK(pij_w, pij_pk, HD, PDIM, PDIM);
  packK(dh_w1, dhw1_pk, PDIM, PDIM, PDIM);
  packK(dh_w2, dhw2_pk, PDIM, NBINS, 80);
  prep_small<<<1,256,0,stream>>>(pb_w, dh_b2, wsum, dh_b2p);

  // -------- embed + pair init
  seq_init<<<(768*HD+255)/256,256,0,stream>>>(x, rp_w, rp_b, pos, hbuf, hbf);
  gemm_wmma<EPI_BIAS><<<dim3(1,12,1),blk,0,stream>>>(hbf,0,HD, pii_pk,0,4, pii_b,
      pib,nullptr,0,PDIM, nullptr,0, HD);
  gemm_wmma<EPI_BIAS><<<dim3(1,12,1),blk,0,stream>>>(hbf,0,HD, pij_pk,0,4, pij_b,
      pjb,nullptr,0,PDIM, nullptr,0, HD);
  pair_init<<<(int)(((long)PROWS*16+255)/256),256,0,stream>>>(pib,pjb,rel_emb,pairb);

  // -------- layers
  for(int i=0;i<NLAY;i++){
    mask_kernel<<<(LSEQ*LSEQ+255)/256,256,0,stream>>>(pairb, wsum+i*PDIM,
        ln_pair_g+i*PDIM, ln_pair_b+i*PDIM, maskb);
    ln_rows<<<192,blk,0,stream>>>(hbuf, ln_seq_g+i*HD, ln_seq_b+i*HD, seqn, HD);
    gemm_wmma<EPI_BIAS><<<dim3(12,12,1),blk,0,stream>>>(seqn,0,HD,
        inw_pk+(long)i*196608,0,48, in_b+i*768, qkvb,nullptr,0,768, nullptr,0, HD);
    attn_pack<<<(8*LSEQ*64+255)/256,256,0,stream>>>(qkvb, qa, kpk, vpk);
    gemm_wmma<EPI_MASK><<<dim3(6,6,8),blk,0,stream>>>(qa,(long)LSEQ*64,64,
        kpk,24576,24, nullptr, scor,nullptr,(long)LSEQ*LSEQ,LSEQ, maskb,LSEQ, 64);
    softmax_rows<<<8*LSEQ,128,0,stream>>>(scor, attn);
    gemm_wmma<EPI_ATTNO><<<dim3(1,6,8),blk,0,stream>>>(attn,(long)LSEQ*LSEQ,LSEQ,
        vpk,24576,4, nullptr, nullptr,obf,0,HD, nullptr,0, LSEQ);
    gemm_wmma<EPI_RES><<<dim3(4,12,1),blk,0,stream>>>(obf,0,HD,
        outw_pk+(long)i*65536,0,16, out_b+i*HD, hbuf,nullptr,0,HD, nullptr,0, HD);
    ln_rows<<<192,blk,0,stream>>>(hbuf, ff_ln_g+i*HD, ff_ln_b+i*HD, ffln, HD);
    gemm_wmma<EPI_GELU_BF><<<dim3(16,12,1),blk,0,stream>>>(ffln,0,HD,
        ffw1_pk+(long)i*262144,0,64, ff_b1+i*1024, nullptr,ffa,0,1024, nullptr,0, HD);
    gemm_wmma<EPI_RES_DUAL><<<dim3(4,12,1),blk,0,stream>>>(ffa,0,1024,
        ffw2_pk+(long)i*262144,0,16, ff_b2+i*HD, hbuf,hbf,0,HD, nullptr,0, 1024);
    gemm_wmma<EPI_BIAS><<<dim3(1,12,1),blk,0,stream>>>(hbf,0,HD,
        outA_pk+(long)i*16384,0,4, outer_b+i*PDIM, oib,nullptr,0,PDIM, nullptr,0, HD);
    gemm_wmma<EPI_BIAS><<<dim3(1,12,1),blk,0,stream>>>(hbf,0,HD,
        outB_pk+(long)i*16384,0,4, nullptr, ojb,nullptr,0,PDIM, nullptr,0, HD);
    pair_fused<0><<<PROWS/64,128,0,stream>>>(pairb, oib, ojb,
        pu_ln_g+i*PDIM, pu_ln_b+i*PDIM,
        puw1_pk+(long)i*4096, pu_b1+i*PDIM,
        puw2_pk+(long)i*4096, pu_b2+i*PDIM, nullptr);
  }

  // -------- distance head + symmetrize
  pair_fused<1><<<PROWS/64,128,0,stream>>>(pairb, nullptr, nullptr,
      dh_ln_g, dh_ln_b, dhw1_pk, dh_b1, dhw2_pk, dh_b2p, logitb);
  symmetrize<<<(int)(((long)PROWS*NBINS+255)/256),256,0,stream>>>(logitb, (float*)d_out);
}
